// MultiHeadAttention_30408368455704
// MI455X (gfx1250) — matrix operand order repaired and checked by reading, not compiled, not run on hardware
//
#include <hip/hip_runtime.h>
#include <hip/hip_bf16.h>
#include <math.h>

typedef __attribute__((ext_vector_type(16))) _Float16 v16h;
typedef __attribute__((ext_vector_type(8)))  float    v8f;

#define NTOK   4096
#define EMBED  20
#define NHEADS 4
#define HDIM   5
#define HID    80     // 4*EMBED
#define KROW   32     // padded Ke row: [f0..15 | 0 x16]
#define SPLITK 4      // key-range split: 4 waves cooperate per (head, q-tile)
// scale folded with log2(e): scores come out of WMMA directly in log2 domain
#define ASCALE (0.44721359549995793f * 1.4426950408889634f)
#define LNEPS  1e-5f

// ---------------------------------------------------------------------------
// prep (runs once):
//  KeA[h][n][32] f16: A-fragment-ready rows [f0..f15, 0x16] with
//    feats = [K | H] (10 real); lane loads v16h at key*32 + hi*16.
//  VeT[h][16][n] f16: transposed V, natural key order; row d=5 is all-ones so
//    the P*V WMMA accumulates the softmax denominator l in output row 5.
// ---------------------------------------------------------------------------
__global__ void prep_kernel(const float* __restrict__ K, const float* __restrict__ V,
                            const float* __restrict__ H,
                            _Float16* __restrict__ KeA, _Float16* __restrict__ VeT) {
  int m = blockIdx.x * blockDim.x + threadIdx.x;
  if (m >= NTOK) return;
  for (int h = 0; h < NHEADS; ++h) {
    _Float16* ke = KeA + ((size_t)(h * NTOK + m)) * KROW;
    // feats: f0..4 = K, f5..9 = H, f10..15 = 0
    for (int i = 0; i < KROW; ++i) ke[i] = (_Float16)0.0f;
    for (int d = 0; d < HDIM; ++d) ke[d] = (_Float16)K[m * EMBED + h * HDIM + d];
    ke[5] = (_Float16)H[m * EMBED + h * HDIM + 0];
    ke[6] = (_Float16)H[m * EMBED + h * HDIM + 1];
    ke[7] = (_Float16)H[m * EMBED + h * HDIM + 2];
    ke[8] = (_Float16)H[m * EMBED + h * HDIM + 3];   // f8
    ke[9] = (_Float16)H[m * EMBED + h * HDIM + 4];   // f9
    for (int d = 0; d < 16; ++d) {
      float v = (d < HDIM) ? V[m * EMBED + h * HDIM + d] : (d == HDIM ? 1.0f : 0.0f);
      VeT[((size_t)(h * 16 + d)) * NTOK + m] = (_Float16)v;
    }
  }
}

// max_m |Ke_m| per head (for the analytic softmax shift bound)
__global__ void norm_kernel(const float* __restrict__ K, const float* __restrict__ H,
                            float* __restrict__ keNorm) {
  int lane = threadIdx.x & 31;
  int h    = threadIdx.x >> 5;   // 4 waves, one per head
  float mx = 0.0f;
  for (int m = lane; m < NTOK; m += 32) {
    float n2 = 0.0f;
    for (int d = 0; d < HDIM; ++d) {
      float a = K[m * EMBED + h * HDIM + d];
      float b = H[m * EMBED + h * HDIM + d];
      n2 += a * a + b * b;
    }
    mx = fmaxf(mx, n2);
  }
  mx = fmaxf(mx, __shfl_xor(mx, 16, 32));
  mx = fmaxf(mx, __shfl_xor(mx, 8, 32));
  mx = fmaxf(mx, __shfl_xor(mx, 4, 32));
  mx = fmaxf(mx, __shfl_xor(mx, 2, 32));
  mx = fmaxf(mx, __shfl_xor(mx, 1, 32));
  if (lane == 0) keNorm[h] = sqrtf(mx);
}

// ---------------------------------------------------------------------------
// One fused transformer layer, all-transposed WMMA with split-K:
//   S^T   = Ke(16k x 32f) x Qe^T(32f x 16q)   -> queries are lanes
//   ctx^T = VeT(16d x 32k) x P^T(32k x 16q)   -> l lands in oacc[5], same lane
// Block = 512 threads = 16 waves = (4 heads) x (4 key splits) per 16-query
// tile; grid = 256 blocks -> 4096 waves device-wide for latency hiding.
// Split partials (fixed shift => plain sums) are combined through LDS.
// ---------------------------------------------------------------------------
__global__ __launch_bounds__(512) void layer_kernel(
    const float* __restrict__ Qin, const float* __restrict__ H,
    const _Float16* __restrict__ KeA, const _Float16* __restrict__ VeT,
    const float* __restrict__ keNorm,
    const float* __restrict__ ln1g, const float* __restrict__ ln1b,
    const float* __restrict__ ln2g, const float* __restrict__ ln2b,
    const float* __restrict__ W1, const float* __restrict__ b1,
    const float* __restrict__ W2, const float* __restrict__ b2,
    float* __restrict__ Qout) {
  __shared__ float cacc[SPLITK][16][NHEADS][6];   // split partials (d0..4, l)
  __shared__ float ctx[16][EMBED];
  __shared__ float q1[16][EMBED];
  __shared__ float hbuf[16][HID];

  const int tid  = threadIdx.x;
  const int lane = tid & 31;
  const int wave = tid >> 5;
  const int head = wave & 3;
  const int sk   = wave >> 2;    // key split 0..3
  const int col  = lane & 15;    // query within tile (lane pair c / c+16)
  const int hi   = lane >> 4;
  const int q0   = blockIdx.x * 16;

  // ---- Qe^T B fragment (built once): lanes 0-15 carry feats 0..15 of their
  // query; lanes 16-31 carry feats 16..31 == 0 ------------------------------
  v16h bq = {};
  float bound;
  {
    const int row = q0 + col;
    const float* qp = Qin + row * EMBED + head * HDIM;
    const float* hp = H   + row * EMBED + head * HDIM;
    float qn2 = 0.0f;
#pragma unroll
    for (int d = 0; d < HDIM; ++d) {
      float a = qp[d], b = hp[d];
      qn2 += a * a + b * b;
      if (hi == 0) {
        bq[d]        = (_Float16)(ASCALE * a);
        bq[HDIM + d] = (_Float16)(ASCALE * b);
      }
    }
    // Cauchy-Schwarz upper bound on the (log2-domain) score, +1 margin.
    // Deterministic per query => identical across split waves => partials add.
    bound = ASCALE * sqrtf(qn2) * keNorm[head] + 1.0f;
  }

  const _Float16* keh = KeA + ((size_t)head * NTOK) * KROW + (size_t)hi * 16;
  const _Float16* vth = VeT + ((size_t)head * 16 + col) * NTOK + (size_t)hi * 8;
  const v8f zc = {};
  v8f oacc = {};

  const int kbeg = sk * (NTOK / SPLITK);
  const int kend = kbeg + (NTOK / SPLITK);
  for (int kb = kbeg; kb < kend; kb += 32) {
    // Ke A fragments: one v16h per lane, zeros for feats 16..31 from memory
    v16h ak0 = *(const v16h*)(keh + (size_t)(kb + col) * KROW);
    v16h ak1 = *(const v16h*)(keh + (size_t)(kb + 16 + col) * KROW);
    v8f s0 = __builtin_amdgcn_wmma_f32_16x16x32_f16(false, ak0, false, bq,
                                                    (short)0, zc, false, false);
    v8f s1 = __builtin_amdgcn_wmma_f32_16x16x32_f16(false, ak1, false, bq,
                                                    (short)0, zc, false, false);

    // p = exp2(s - bound) in (0,1]; raw v_exp_f32, per-lane, no reductions.
    // With the middle-group K permutation, P^T B-frag is just [p0 | p1].
    v16h bp;
#pragma unroll
    for (int r = 0; r < 8; ++r) {
      bp[r]     = (_Float16)__builtin_amdgcn_exp2f(s0[r] - bound);
      bp[8 + r] = (_Float16)__builtin_amdgcn_exp2f(s1[r] - bound);
    }

    // VeT A fragment: keys kb+hi*8 .. +7 and kb+16+hi*8 .. +7
    v16h av;
#pragma unroll
    for (int s = 0; s < 8; ++s) { av[s] = vth[kb + s]; av[8 + s] = vth[kb + 16 + s]; }

    oacc = __builtin_amdgcn_wmma_f32_16x16x32_f16(false, av, false, bp,
                                                  (short)0, oacc, false, false);
  }

  // ---- store split partials (rows = dims 0..4 + l in row 5) --------------
  if (hi == 0) {
#pragma unroll
    for (int d = 0; d < 6; ++d) cacc[sk][col][head][d] = oacc[d];
  }
  __syncthreads();

  // ---- combine splits + normalize: 64 threads = (row, head) --------------
  if (tid < 64) {
    int row = tid >> 2, h = tid & 3;
    float sum[6];
#pragma unroll
    for (int d = 0; d < 6; ++d)
      sum[d] = cacc[0][row][h][d] + cacc[1][row][h][d] +
               cacc[2][row][h][d] + cacc[3][row][h][d];
    float inv = 1.0f / sum[HDIM];
#pragma unroll
    for (int d = 0; d < HDIM; ++d)
      ctx[row][h * HDIM + d] = sum[d] * inv;
  }
  __syncthreads();

  // ---- residual + LN1 (row-local, E=20) ---------------------------------
  if (tid < 16) {
    int g = q0 + tid;
    float x[EMBED]; float mu = 0.0f;
#pragma unroll
    for (int j = 0; j < EMBED; ++j) { x[j] = Qin[g * EMBED + j] + ctx[tid][j]; mu += x[j]; }
    mu *= (1.0f / EMBED);
    float var = 0.0f;
#pragma unroll
    for (int j = 0; j < EMBED; ++j) { float d = x[j] - mu; var += d * d; }
    var *= (1.0f / EMBED);
    float inv = rsqrtf(var + LNEPS);
#pragma unroll
    for (int j = 0; j < EMBED; ++j)
      q1[tid][j] = (x[j] - mu) * inv * ln1g[j] + ln1b[j];
  }
  __syncthreads();

  // ---- FFN hidden: 128 threads = 16 rows x 8, 10 hidden units each ------
  if (tid < 128) {
    int row = tid >> 3;
    int j0  = (tid & 7) * 10;
    float xr[EMBED];
#pragma unroll
    for (int k = 0; k < EMBED; ++k) xr[k] = q1[row][k];
    for (int jj = 0; jj < 10; ++jj) {
      int j = j0 + jj;
      float acc = b1[j];
#pragma unroll
      for (int k = 0; k < EMBED; ++k) acc += xr[k] * W1[k * HID + j];
      hbuf[row][j] = tanhf(acc);
    }
  }
  __syncthreads();

  // ---- FFN out + residual -----------------------------------------------
  if (tid < 128) {
    int row = tid >> 3;
    for (int j = tid & 7; j < EMBED; j += 8) {
      float acc = b2[j];
#pragma unroll
      for (int k = 0; k < HID; ++k) acc += hbuf[row][k] * W2[k * EMBED + j];
      ctx[row][j] = q1[row][j] + acc;   // reuse ctx as y buffer
    }
  }
  __syncthreads();

  // ---- LN2 -> Qout -------------------------------------------------------
  if (tid < 16) {
    int g = q0 + tid;
    float mu = 0.0f;
#pragma unroll
    for (int j = 0; j < EMBED; ++j) mu += ctx[tid][j];
    mu *= (1.0f / EMBED);
    float var = 0.0f;
#pragma unroll
    for (int j = 0; j < EMBED; ++j) { float d = ctx[tid][j] - mu; var += d * d; }
    var *= (1.0f / EMBED);
    float inv = rsqrtf(var + LNEPS);
#pragma unroll
    for (int j = 0; j < EMBED; ++j)
      Qout[g * EMBED + j] = (ctx[tid][j] - mu) * inv * ln2g[j] + ln2b[j];
  }
}

// ---------------------------------------------------------------------------
extern "C" void kernel_launch(void* const* d_in, const int* in_sizes, int n_in,
                              void* d_out, int out_size, void* d_ws, size_t ws_size,
                              hipStream_t stream) {
  (void)in_sizes; (void)n_in; (void)out_size; (void)ws_size;
  const float* Q    = (const float*)d_in[0];
  const float* K    = (const float*)d_in[1];
  const float* V    = (const float*)d_in[2];
  const float* H    = (const float*)d_in[3];
  const float* ln1g = (const float*)d_in[4];
  const float* ln1b = (const float*)d_in[5];
  const float* ln2g = (const float*)d_in[6];
  const float* ln2b = (const float*)d_in[7];
  const float* W1   = (const float*)d_in[8];
  const float* b1   = (const float*)d_in[9];
  const float* W2   = (const float*)d_in[10];
  const float* b2   = (const float*)d_in[11];

  _Float16* KeA = (_Float16*)d_ws;                                  // 1 MB
  _Float16* VeT = KeA + (size_t)NHEADS * NTOK * KROW;               // 512 KB
  float*    Qbuf   = (float*)(VeT + (size_t)NHEADS * 16 * NTOK);    // 320 KB
  float*    keNorm = Qbuf + (size_t)NTOK * EMBED;                   // 4 floats
  float*    Qfin = (float*)d_out;

  prep_kernel<<<NTOK / 256, 256, 0, stream>>>(K, V, H, KeA, VeT);
  norm_kernel<<<1, 128, 0, stream>>>(K, H, keNorm);

  layer_kernel<<<NTOK / 16, 512, 0, stream>>>(Q,    H, KeA, VeT, keNorm, ln1g, ln1b, ln2g, ln2b, W1, b1, W2, b2, Qbuf);
  layer_kernel<<<NTOK / 16, 512, 0, stream>>>(Qbuf, H, KeA, VeT, keNorm, ln1g, ln1b, ln2g, ln2b, W1, b1, W2, b2, Qfin);
  layer_kernel<<<NTOK / 16, 512, 0, stream>>>(Qfin, H, KeA, VeT, keNorm, ln1g, ln1b, ln2g, ln2b, W1, b1, W2, b2, Qbuf);
  layer_kernel<<<NTOK / 16, 512, 0, stream>>>(Qbuf, H, KeA, VeT, keNorm, ln1g, ln1b, ln2g, ln2b, W1, b1, W2, b2, Qfin);
}
